// TipAdapterLinear_67422396612889
// MI455X (gfx1250) — compile-verified, matrix operand order checked by reading, not run on hardware
//
#include <hip/hip_runtime.h>
#include <hip/hip_bf16.h>

typedef __attribute__((ext_vector_type(16))) __bf16 v16bf;
typedef __attribute__((ext_vector_type(8)))  __bf16 v8bf;
typedef __attribute__((ext_vector_type(8)))  float  v8f;

#define D_DIM  1024   // feature dim (fixed by reference)
#define N_DIM  16384  // cache entries (fixed by reference)
#define BM     16     // query rows per workgroup
#define WAVES  8
#define NC     128    // N-chunk (16 columns per wave, 8 waves)
#define DT     8      // 16x16 output d-tiles per wave (128 cols * 8 waves = 1024)
#define PSTR   (NC + 8)   // P row stride in ushorts: 272B = 68 dwords -> 4-bank skew

__device__ __forceinline__ unsigned short f2bf(float f) {
  union { float f; unsigned u; } v; v.f = f;
  unsigned r = v.u + 0x7FFFu + ((v.u >> 16) & 1u);   // round-to-nearest-even
  return (unsigned short)(r >> 16);
}

// ---------------------------------------------------------------------------
// Prep: L2-normalize each 1024-wide row, optional scale (beta*log2e), emit bf16
// ---------------------------------------------------------------------------
__global__ __launch_bounds__(256)
void prep_rows(const float* __restrict__ src, unsigned short* __restrict__ dst,
               const float* __restrict__ scale_ptr)
{
  __shared__ float red[8];
  const int row = blockIdx.x;
  const float* p = src + (size_t)row * D_DIM;
  float s = 0.f;
  for (int i = threadIdx.x; i < D_DIM; i += 256) { float v = p[i]; s += v * v; }
  s += __shfl_xor(s, 1);  s += __shfl_xor(s, 2);  s += __shfl_xor(s, 4);
  s += __shfl_xor(s, 8);  s += __shfl_xor(s, 16);
  if ((threadIdx.x & 31) == 0) red[threadIdx.x >> 5] = s;
  __syncthreads();
  if (threadIdx.x == 0) {
    float t = 0.f;
    for (int i = 0; i < 8; ++i) t += red[i];
    red[0] = t;
  }
  __syncthreads();
  const float scale = scale_ptr ? (scale_ptr[0] * 1.44269504088896340736f) : 1.0f;
  const float rn = scale / fmaxf(sqrtf(red[0]), 1e-12f);
  unsigned short* q = dst + (size_t)row * D_DIM;
  for (int i = threadIdx.x; i < D_DIM; i += 256) q[i] = f2bf(p[i] * rn);
}

// ---------------------------------------------------------------------------
// Prep: L2-normalize rows of V [N,1024] and store transposed bf16 Vt [1024,N]
// ---------------------------------------------------------------------------
__global__ __launch_bounds__(256)
void prep_vt(const float* __restrict__ v, unsigned short* __restrict__ vt)
{
  __shared__ float part[16][17];
  __shared__ float rn[16];
  const int n0 = blockIdx.x * 16;
  const int tr = threadIdx.x >> 4;   // 0..15
  const int tc = threadIdx.x & 15;   // 0..15
  const float* p = v + (size_t)(n0 + tr) * D_DIM;
  float s = 0.f;
  for (int j = tc; j < D_DIM; j += 16) { float x = p[j]; s += x * x; }
  part[tr][tc] = s;
  __syncthreads();
  if (threadIdx.x < 16) {
    float t = 0.f;
    for (int i = 0; i < 16; ++i) t += part[threadIdx.x][i];
    rn[threadIdx.x] = 1.f / fmaxf(sqrtf(t), 1e-12f);
  }
  __syncthreads();
  for (int db = 0; db < D_DIM / 16; ++db) {
    const int d = db * 16 + tr;
    const float val = v[(size_t)(n0 + tc) * D_DIM + d] * rn[tc];
    vt[(size_t)d * N_DIM + n0 + tc] = f2bf(val);  // 16 contiguous bf16 / 16 thr
  }
}

// ---------------------------------------------------------------------------
// Fused flash-softmax double GEMM.
//   grid = B/16 blocks, 256 threads (8 waves of 32).
//   Wave w: S columns [16w,16w+16) per chunk; output d-cols [128w, 128w+128).
//   All intra-chunk load displacements are compile-time immediates on
//   per-chunk-invariant pointers (fit signed 24-bit IOFFSET).
//   launch_bounds(256,4): force <=256 VGPRs (no MSB toggles) + 2 blocks/WGP.
// ---------------------------------------------------------------------------
__global__ __launch_bounds__(256, 4)
void tip_main(const float* __restrict__ x,
              const unsigned short* __restrict__ qb,   // [B,1024] bf16, *beta*log2e
              const unsigned short* __restrict__ kb,   // [N,1024] bf16
              const unsigned short* __restrict__ vt,   // [1024,N] bf16
              const float* __restrict__ alpha_p,
              float* __restrict__ out)
{
  __shared__ __align__(32) unsigned short P[BM][PSTR];   // exp weights, bf16
  __shared__ float wavemax[WAVES][BM];
  __shared__ float wavesum[WAVES][BM];
  __shared__ float m_run[BM], l_run[BM], corr_s[BM], mnew_s[BM];

  const int tid  = threadIdx.x;
  const int w    = tid >> 5;        // wave id 0..7
  const int lane = tid & 31;
  const int lh   = lane >> 4;       // half-wave (K-half select per ISA layout)
  const int l16  = lane & 15;
  const int m0   = blockIdx.x * BM;

  v8f acc[DT];
#pragma unroll
  for (int t = 0; t < DT; ++t)
    acc[t] = (v8f){0.f, 0.f, 0.f, 0.f, 0.f, 0.f, 0.f, 0.f};

  if (tid < BM) { m_run[tid] = -__builtin_huge_valf(); l_run[tid] = 0.f; }
  __syncthreads();

  const int dc0 = w * (DT * 16) + l16;                 // wave's first d col
  // chunk-invariant / running pointers (constant immediates handle the rest)
  const unsigned short* pq = qb + (size_t)(m0 + l16) * D_DIM + lh * 16;
  const unsigned short* pk = kb + (size_t)(w * 16 + l16) * D_DIM + lh * 16;
  const unsigned short* pv = vt + (size_t)dc0 * N_DIM + lh * 16;

  for (int nb = 0; nb < N_DIM; nb += NC) {
    // ---- S tile: q[16,1024] x k[16 cols,1024]^T, f32 acc, double-buffered ----
    v8f Sa = (v8f){0.f, 0.f, 0.f, 0.f, 0.f, 0.f, 0.f, 0.f};
    v8f Sb = Sa;
    v16bf a_cur = *(const v16bf*)(pq);
    v16bf b_cur = *(const v16bf*)(pk);
#pragma unroll
    for (int kk = 0; kk < D_DIM / 32; ++kk) {
      const int nxt = (kk + 1 < D_DIM / 32) ? (kk + 1) : kk;
      v16bf a_nxt = *(const v16bf*)(pq + nxt * 32);    // imm offset
      v16bf b_nxt = *(const v16bf*)(pk + nxt * 32);    // imm offset
      if (kk & 1)
        Sb = __builtin_amdgcn_wmma_f32_16x16x32_bf16(false, a_cur, false, b_cur,
                                                     (short)0, Sb, false, false);
      else
        Sa = __builtin_amdgcn_wmma_f32_16x16x32_bf16(false, a_cur, false, b_cur,
                                                     (short)0, Sa, false, false);
      a_cur = a_nxt; b_cur = b_nxt;
    }
    v8f S = Sa + Sb;

    // ---- per-row max over this wave's 16 columns (16-lane groups) ----
    float pm[8];
#pragma unroll
    for (int r = 0; r < 8; ++r) {
      float m = S[r];
      m = fmaxf(m, __shfl_xor(m, 1));
      m = fmaxf(m, __shfl_xor(m, 2));
      m = fmaxf(m, __shfl_xor(m, 4));
      m = fmaxf(m, __shfl_xor(m, 8));
      pm[r] = m;
    }
    if (l16 == 0) {
#pragma unroll
      for (int r = 0; r < 8; ++r) wavemax[w][lh * 8 + r] = pm[r];
    }
    __syncthreads();                                   // (A)

    // ---- wave0: online-softmax stats ----
    if (tid < BM) {
      float mc = wavemax[0][tid];
#pragma unroll
      for (int i = 1; i < WAVES; ++i) mc = fmaxf(mc, wavemax[i][tid]);
      const float mo = m_run[tid];
      const float mn = fmaxf(mo, mc);
      m_run[tid]  = mn;
      mnew_s[tid] = mn;
      corr_s[tid] = exp2f(mo - mn);                    // 0 on first chunk
    }
    __syncthreads();                                   // (B)

    float cr[8], mnv[8];
#pragma unroll
    for (int r = 0; r < 8; ++r) {
      cr[r]  = corr_s[lh * 8 + r];                     // row = r + 8*lh
      mnv[r] = mnew_s[lh * 8 + r];
    }
    const float csum = cr[0]+cr[1]+cr[2]+cr[3]+cr[4]+cr[5]+cr[6]+cr[7];
    if (__any(csum != 8.0f)) {                         // rescale only if max moved
#pragma unroll
      for (int t = 0; t < DT; ++t)
#pragma unroll
        for (int r = 0; r < 8; ++r) acc[t][r] *= cr[r];
    }

    // ---- P = exp2(S - m), store bf16 + partial row sums ----
    float ps[8];
#pragma unroll
    for (int r = 0; r < 8; ++r) {
      const float p = exp2f(S[r] - mnv[r]);
      ps[r] = p;
      P[lh * 8 + r][w * 16 + l16] = f2bf(p);
    }
#pragma unroll
    for (int r = 0; r < 8; ++r) {
      ps[r] += __shfl_xor(ps[r], 1);
      ps[r] += __shfl_xor(ps[r], 2);
      ps[r] += __shfl_xor(ps[r], 4);
      ps[r] += __shfl_xor(ps[r], 8);
    }
    if (l16 == 0) {
#pragma unroll
      for (int r = 0; r < 8; ++r) wavesum[w][lh * 8 + r] = ps[r];
    }
    __syncthreads();                                   // (C)

    if (tid < BM) {
      float s = wavesum[0][tid];
#pragma unroll
      for (int i = 1; i < WAVES; ++i) s += wavesum[i][tid];
      l_run[tid] = l_run[tid] * corr_s[tid] + s;
    }

    // ---- acc += P[16,128] x V[128, wave's 128 cols] ----
    // All 4 A-fragments from LDS up front; B double-buffered from Vt (L2-hot),
    // every displacement a compile-time immediate on pv.
    v16bf apv[NC / 32];
#pragma unroll
    for (int kk2 = 0; kk2 < NC / 32; ++kk2) {
      v8bf plo = *(const v8bf*)(&P[l16][kk2 * 32 + lh * 16]);
      v8bf phi = *(const v8bf*)(&P[l16][kk2 * 32 + lh * 16 + 8]);
      apv[kk2] = __builtin_shufflevector(plo, phi,
                   0, 1, 2, 3, 4, 5, 6, 7, 8, 9, 10, 11, 12, 13, 14, 15);
    }
    v16bf b_pv = *(const v16bf*)(pv);                  // i = 0
#pragma unroll
    for (int kk2 = 0; kk2 < NC / 32; ++kk2) {
#pragma unroll
      for (int t = 0; t < DT; ++t) {
        const int i  = kk2 * DT + t;
        const int in = (i + 1 < (NC / 32) * DT) ? (i + 1) : i;
        const int voff = (in / DT) * 32 + (in % DT) * 16 * N_DIM;  // const
        v16bf b_nxt = *(const v16bf*)(pv + voff);      // imm offset (<= 3.7MB)
        acc[t] = __builtin_amdgcn_wmma_f32_16x16x32_bf16(false, apv[kk2], false,
                                                         b_pv, (short)0, acc[t],
                                                         false, false);
        b_pv = b_nxt;
      }
    }

    pk += (size_t)NC * D_DIM;                          // next chunk's key rows
    pv += NC;                                          // next chunk's Vt cols
  }

  __syncthreads();
  const float alpha = alpha_p[0];
  float li[8];
#pragma unroll
  for (int r = 0; r < 8; ++r) li[r] = alpha / l_run[lh * 8 + r];

#pragma unroll
  for (int t = 0; t < DT; ++t) {
    const unsigned dcol = (unsigned)(dc0 + t * 16);
#pragma unroll
    for (int r = 0; r < 8; ++r) {
      const unsigned idx = (unsigned)(m0 + lh * 8 + r) * D_DIM + dcol;
      out[idx] = __builtin_fmaf(acc[t][r], li[r], x[idx]);
    }
  }
}

// ---------------------------------------------------------------------------
extern "C" void kernel_launch(void* const* d_in, const int* in_sizes, int n_in,
                              void* d_out, int out_size, void* d_ws, size_t ws_size,
                              hipStream_t stream)
{
  const float* x     = (const float*)d_in[0];
  const float* ck    = (const float*)d_in[1];
  const float* cv    = (const float*)d_in[2];
  const float* beta  = (const float*)d_in[3];
  const float* alpha = (const float*)d_in[4];
  float* out = (float*)d_out;

  const int B  = in_sizes[0] / D_DIM;   // 8192
  const int Nn = in_sizes[1] / D_DIM;   // 16384 (== N_DIM)

  unsigned short* qb = (unsigned short*)d_ws;             // B*1024 bf16 (16 MB)
  unsigned short* kb = qb + (size_t)B * D_DIM;            // N*1024 bf16 (32 MB)
  unsigned short* vt = kb + (size_t)Nn * D_DIM;           // 1024*N bf16 (32 MB)

  prep_rows<<<B, 256, 0, stream>>>(x, qb, beta);          // q: fold beta*log2e
  prep_rows<<<Nn, 256, 0, stream>>>(ck, kb, nullptr);     // keys
  prep_vt<<<Nn / 16, 256, 0, stream>>>(cv, vt);           // values, transposed

  tip_main<<<B / BM, 256, 0, stream>>>(x, qb, kb, vt, alpha, out);
}
